// HeadQK_62843961475103
// MI455X (gfx1250) — matrix operand order repaired and checked by reading, not compiled, not run on hardware
//
#include <hip/hip_runtime.h>
#include <hip/hip_bf16.h>

typedef __attribute__((ext_vector_type(16))) _Float16 v16h;
typedef __attribute__((ext_vector_type(8)))  _Float16 v8h;
typedef __attribute__((ext_vector_type(8)))  float    v8f;

#define TILE 16

// ---------------------------------------------------------------------------
// Kernel 1: zero-fill the dense [T, V] fp32 output (524 MB) with float4 stores.
// ---------------------------------------------------------------------------
__global__ void zero_out_kernel(float4* __restrict__ out, long long n4) {
  long long i = (long long)blockIdx.x * blockDim.x + threadIdx.x;
  long long stride = (long long)gridDim.x * blockDim.x;
  float4 z = make_float4(0.f, 0.f, 0.f, 0.f);
  for (; i < n4; i += stride) out[i] = z;
}

// ---------------------------------------------------------------------------
// Kernel 2: Q/K projection.  q = x@Wq, k = x@Wk, stored f16 into workspace.
// One wave computes one 16x16 output tile; 8 waves per block.
// grid = (T/16, D/16/8, 2), block = 256 (8 waves).
// fp32 inputs converted to f16 on the fly; accumulate in f32 via WMMA.
// ---------------------------------------------------------------------------
__global__ __launch_bounds__(256) void qk_proj_kernel(
    const float* __restrict__ x, const float* __restrict__ Wq,
    const float* __restrict__ Wk, _Float16* __restrict__ qk,
    int T, int C, int D) {
  const int lane  = threadIdx.x & 31;
  const int wave  = threadIdx.x >> 5;
  const int mTile = blockIdx.x;
  const int nTile = blockIdx.y * 8 + wave;
  const float* __restrict__ W = (blockIdx.z == 0) ? Wq : Wk;
  _Float16* __restrict__ out = qk + (size_t)blockIdx.z * (size_t)T * D;

  const int mRow   = mTile * TILE + (lane & 15);  // A row this lane feeds
  const int nCol   = nTile * TILE + (lane & 15);  // B col this lane feeds
  const int kHalfA = (lane >> 4) * 8;             // A K-base (ISA layout)
  const int kBaseB = (lane >> 4) * 8;             // B K-base (ISA layout)

  v8f acc = {};
  for (int k0 = 0; k0 < C; k0 += 32) {
    // A: x[mRow, k0+kHalfA+{0..7}] and +16..23  (row-major contiguous)
    const float* arow = x + (size_t)mRow * C + k0 + kHalfA;
    __builtin_prefetch(arow + 32, 0, 0);  // next K-step -> global_prefetch_b8
    v16h a, b;
#pragma unroll
    for (int i = 0; i < 8; ++i) {
      a[i]     = (_Float16)arow[i];
      a[i + 8] = (_Float16)arow[i + 16];
    }
    // B: W[k0+kBaseB+{0..7}, nCol] and +16..23 (strided column reads, L2-resident)
    const float* bptr = W + (size_t)(k0 + kBaseB) * D + nCol;
#pragma unroll
    for (int i = 0; i < 16; ++i) b[i] = (_Float16)bptr[(size_t)(i < 8 ? i : i + 8) * D];

    acc = __builtin_amdgcn_wmma_f32_16x16x32_f16(false, a, false, b,
                                                 (short)0, acc, false, false);
  }

  // C/D layout: VGPR r -> row (lane/16)*8 + r, col lane%16
  const int rBase = (lane >> 4) * 8;
#pragma unroll
  for (int r = 0; r < 8; ++r) {
    int row = mTile * TILE + rBase + r;
    out[(size_t)row * D + nTile * TILE + (lane & 15)] = (_Float16)acc[r];
  }
}

// ---------------------------------------------------------------------------
// Kernel 3: scores + causal mask + vocab scatter.
// One wave computes a 16x16 tile of c = (q@k^T)/256, then atomically
// scatter-adds it into out[t, idx[j]].  Tiles fully above the diagonal are
// skipped (halves compute and atomic traffic).  unsafeAtomicAdd guarantees
// the hardware global_atomic_add_f32 (non-returning, fire-and-forget) rather
// than a CAS loop.
// grid = (T/16, T/16/8), block = 256 (8 waves).
// ---------------------------------------------------------------------------
__global__ __launch_bounds__(256) void attn_scatter_kernel(
    const _Float16* __restrict__ q, const _Float16* __restrict__ kmat,
    const int* __restrict__ idx, float* __restrict__ out,
    int T, int D, int V) {
  const int lane  = threadIdx.x & 31;
  const int wave  = threadIdx.x >> 5;
  const int iTile = blockIdx.x;
  const int jTile = blockIdx.y * 8 + wave;
  if (jTile > iTile) return;  // fully masked tile

  const int mRow   = iTile * TILE + (lane & 15);  // q row
  const int nRow   = jTile * TILE + (lane & 15);  // k row (B = k^T column)
  const int kHalfA = (lane >> 4) * 8;
  const int kBaseB = (lane >> 4) * 8;

  v8f acc = {};
#pragma unroll
  for (int k0 = 0; k0 < 256; k0 += 32) {
    const _Float16* arow = q    + (size_t)mRow * D + k0 + kHalfA;
    const _Float16* brow = kmat + (size_t)nRow * D + k0 + kBaseB;
    v8h a0 = *(const v8h*)arow;
    v8h a1 = *(const v8h*)(arow + 16);
    v8h b0 = *(const v8h*)brow;
    v8h b1 = *(const v8h*)(brow + 16);
    v16h a, b;
#pragma unroll
    for (int i = 0; i < 8; ++i) {
      a[i] = a0[i]; a[i + 8] = a1[i];
      b[i] = b0[i]; b[i + 8] = b1[i];
    }
    acc = __builtin_amdgcn_wmma_f32_16x16x32_f16(false, a, false, b,
                                                 (short)0, acc, false, false);
  }

  const float scale = 1.0f / 256.0f;
  const int jCol  = jTile * TILE + (lane & 15);  // this lane's column j
  const int v     = idx[jCol];                   // vocab slot (one load, 8 uses)
  const int rBase = (lane >> 4) * 8;
#pragma unroll
  for (int r = 0; r < 8; ++r) {
    int t = iTile * TILE + rBase + r;
    if (t >= jCol)  // causal (tril includes diagonal)
      (void)unsafeAtomicAdd(out + (size_t)t * V + v, acc[r] * scale);
  }
}

// ---------------------------------------------------------------------------
extern "C" void kernel_launch(void* const* d_in, const int* in_sizes, int n_in,
                              void* d_out, int out_size, void* d_ws, size_t ws_size,
                              hipStream_t stream) {
  const float* x   = (const float*)d_in[0];
  const int*   idx = (const int*)d_in[1];
  const float* Wq  = (const float*)d_in[2];
  const float* Wk  = (const float*)d_in[3];
  float* out = (float*)d_out;

  const int T = in_sizes[1];            // 4096
  const int C = in_sizes[0] / T;        // 1024
  const int D = in_sizes[2] / C;        // 256
  const int V = (int)((long long)out_size / T);  // 32000

  _Float16* qk = (_Float16*)d_ws;       // q at [0], k at [T*D]; 4 MB total

  long long n4 = (long long)out_size / 4;
  zero_out_kernel<<<4096, 256, 0, stream>>>((float4*)out, n4);

  dim3 gProj(T / TILE, (D / TILE) / 8, 2);
  qk_proj_kernel<<<gProj, 256, 0, stream>>>(x, Wq, Wk, qk, T, C, D);

  dim3 gAttn(T / TILE, (T / TILE) / 8);
  attn_scatter_kernel<<<gAttn, 256, 0, stream>>>(qk, qk + (size_t)T * D, idx,
                                                 out, T, D, V);
}
